// HybridQLSTM_65481071397065
// MI455X (gfx1250) — matrix operand order repaired and checked by reading, not compiled, not run on hardware
//
#include <hip/hip_runtime.h>

// ---------------------------------------------------------------------------
// QLSTM for MI455X (gfx1250, wave32, WMMA).
//   K0 setup : build bf16 B-layout tables for Wx/Wh, fold bias+theta, zero the
//              per-timestep producer counters.
//   K1 fused : blocks 0..31   = recurrence consumers (1 wave = 16 batch rows,
//                               chases producers via acquire-spin on cnt[s]);
//              blocks 32..+32K = Zx producers: Zx = x @ Wx.T + (b+theta) via
//                               v_wmma_f32_16x16x32_bf16, release-increment
//                               cnt[s] when a tile is written.
//              Consumer step:  D_g = WMMA(h_bf16, Wh_g, C = Zx tile) ->
//                              cos/sigmoid/tanh -> c,h update, h transposed
//                              D->A layout via 512 B of LDS.
//   K2 head  : hmean @ Wc.T + bc, log_softmax.
// ---------------------------------------------------------------------------

typedef __bf16 bf16_t;
typedef __attribute__((ext_vector_type(16))) bf16_t v16bf;
typedef __attribute__((ext_vector_type(8)))  float  v8f;

#define SEQ    1024
#define BATCH  512
#define DIN    128
#define HID    16
#define NTILE  32                          // BATCH / 16
#define NPROD  (SEQ * NTILE)               // producer blocks
#define ZX_FLOATS (SEQ * BATCH * 64)       // 33554432 floats
#define ZX_BYTES  ((size_t)ZX_FLOATS * 4)  // 128 MB
#define WT_ELEMS  (4 * 4 * 32 * 16)        // Wx table, bf16
#define WH_ELEMS  (4 * 32 * 16)            // Wh table, bf16
#define SSTRIDE   (NTILE * 4 * 256)        // Zx float stride per timestep

__device__ __forceinline__ float fast_cos(float x) {
  // v_cos_f32 computes cos(2*pi*x)
  return __builtin_amdgcn_cosf(x * 0.15915494309189535f);
}
__device__ __forceinline__ float fast_sigmoid(float x) {
  return __builtin_amdgcn_rcpf(1.0f + __builtin_amdgcn_exp2f(x * -1.4426950408889634f));
}
__device__ __forceinline__ float fast_tanh(float x) {
#if __has_builtin(__builtin_amdgcn_tanhf)
  return __builtin_amdgcn_tanhf(x);
#else
  float e = __builtin_amdgcn_exp2f(x * 2.885390081777927f);  // 2^(2x*log2 e)
  return 1.0f - 2.0f * __builtin_amdgcn_rcpf(e + 1.0f);
#endif
}
__device__ __forceinline__ int acq_load(const int* p) {
  return __hip_atomic_load(p, __ATOMIC_ACQUIRE, __HIP_MEMORY_SCOPE_AGENT);
}

// ------------------------------ K0: setup ----------------------------------
__global__ void qlstm_setup(const float* Wf, const float* bf_, const float* thf,
                            const float* Wi, const float* bi_, const float* thi,
                            const float* Wu, const float* bu_, const float* thu,
                            const float* Wo, const float* bo_, const float* tho,
                            bf16_t* __restrict__ wtab, bf16_t* __restrict__ whtab,
                            float* __restrict__ be, int* __restrict__ cnt) {
  const float* W[4]  = {Wf, Wi, Wu, Wo};
  const float* bb[4] = {bf_, bi_, bu_, bo_};
  const float* th[4] = {thf, thi, thu, tho};
  const int t = threadIdx.x + blockIdx.x * blockDim.x;
  const int stride = blockDim.x * gridDim.x;

  // Wx in B-layout bf16: element(k,n): lane = n + 16*((k/8)%2), elem = (k%8) + 8*(k/16).
  for (int idx = t; idx < WT_ELEMS; idx += stride) {
    int i = idx & 15, lane = (idx >> 4) & 31, cch = (idx >> 9) & 3, g = idx >> 11;
    int n = lane & 15;
    int kk = cch * 32 + ((lane >> 4) * 8) + (i & 7) + ((i >> 3) * 16);          // 0..127
    wtab[idx] = (bf16_t)W[g][n * (DIN + HID) + kk];
  }
  // Wh in B-layout, K=0..15 real (elems 0..7, K = 8*(lane/16)+elem), K=16..31 zero (elems 8..15 zero).
  for (int idx = t; idx < WH_ELEMS; idx += stride) {
    int i = idx & 15, lane = (idx >> 4) & 31, g = idx >> 9;
    float v = (i < 8) ? W[g][(lane & 15) * (DIN + HID) + DIN + (lane >> 4) * 8 + i] : 0.0f;
    whtab[idx] = (bf16_t)v;
  }
  // effective bias = b + theta (cos applied to z + theta)
  for (int idx = t; idx < 64; idx += stride) {
    int g = idx >> 4, j = idx & 15;
    be[idx] = bb[g][j] + th[g][j];
  }
  // producer-progress counters (must be re-zeroed every call)
  for (int idx = t; idx < SEQ; idx += stride) cnt[idx] = 0;
}

// ------------------------------ K1: fused producer/consumer ----------------
__global__ void __launch_bounds__(32) qlstm_fused(const float* __restrict__ x,
                                                  const bf16_t* __restrict__ wtab,
                                                  const bf16_t* __restrict__ whtab,
                                                  const float* __restrict__ be,
                                                  float* __restrict__ zx,
                                                  int* __restrict__ cnt,
                                                  float* __restrict__ hmean) {
  const int lane = threadIdx.x;
  const int n    = lane & 15;
  const int hsel = lane >> 4;

  if ((int)blockIdx.x >= NTILE) {
    // ---------------- producer: one 16-row tile of Zx at timestep s --------
    const int pbid = (int)blockIdx.x - NTILE;
    const int s    = pbid >> 5;
    const int tile = pbid & 31;

    const int b = tile * 16 + n;                  // A row M = lane&15
    const float* xr = x + ((size_t)b * SEQ + s) * DIN;

    v8f acc[4];
#pragma unroll
    for (int g = 0; g < 4; ++g) {
      float bv = be[g * 16 + n];
      acc[g] = (v8f){bv, bv, bv, bv, bv, bv, bv, bv};
    }

#pragma unroll
    for (int cch = 0; cch < 4; ++cch) {
      // A layout (16-bit 16x32): lanes 0-15 K=0..7/16..23, lanes 16-31 K=8..15/24..31
      const int d0 = cch * 32 + hsel * 8;
      float4 A0 = *(const float4*)(xr + d0);
      float4 A1 = *(const float4*)(xr + d0 + 4);
      float4 A2 = *(const float4*)(xr + d0 + 16);
      float4 A3 = *(const float4*)(xr + d0 + 20);
      v16bf a;
      a[0]  = (bf16_t)A0.x; a[1]  = (bf16_t)A0.y; a[2]  = (bf16_t)A0.z; a[3]  = (bf16_t)A0.w;
      a[4]  = (bf16_t)A1.x; a[5]  = (bf16_t)A1.y; a[6]  = (bf16_t)A1.z; a[7]  = (bf16_t)A1.w;
      a[8]  = (bf16_t)A2.x; a[9]  = (bf16_t)A2.y; a[10] = (bf16_t)A2.z; a[11] = (bf16_t)A2.w;
      a[12] = (bf16_t)A3.x; a[13] = (bf16_t)A3.y; a[14] = (bf16_t)A3.z; a[15] = (bf16_t)A3.w;
#pragma unroll
      for (int g = 0; g < 4; ++g) {
        v16bf wb = *(const v16bf*)(wtab + ((size_t)((g * 4 + cch) * 32 + lane) * 16));
        acc[g] = __builtin_amdgcn_wmma_f32_16x16x32_bf16(false, a, false, wb,
                                                         (short)0, acc[g], false, false);
      }
    }

    // Swizzled store matching WMMA C/D layout: [s][tile][gate][lane][8]
    float* out = zx + ((size_t)(s * NTILE + tile) * 4) * 256 + (size_t)lane * 8;
#pragma unroll
    for (int g = 0; g < 4; ++g) *(v8f*)(out + g * 256) = acc[g];

    __threadfence();                      // make tile visible at agent scope
    if (lane == 0) atomicAdd(&cnt[s], 1); // announce tile completion
    return;
  }

  // ------------------ consumer: recurrence over S for one 16-row tile ------
  const int tile = (int)blockIdx.x;

  __shared__ bf16_t hlds[256];  // 16x16 row-major bf16

  // Wh B operands (resident in VGPRs for the whole sequence)
  v16bf whb[4];
#pragma unroll
  for (int g = 0; g < 4; ++g)
    whb[g] = *(const v16bf*)(whtab + (size_t)(g * 32 + lane) * 16);

  // h in A layout (bf16), zero-initialized; elements 8..15 (K=16..31) stay 0.
  union AOp { v16bf v; float4 f4[2]; } ah;
  ah.f4[0] = make_float4(0.f, 0.f, 0.f, 0.f);
  ah.f4[1] = make_float4(0.f, 0.f, 0.f, 0.f);

  v8f c    = (v8f){0.f, 0.f, 0.f, 0.f, 0.f, 0.f, 0.f, 0.f};
  v8f hsum = c;

  const float* zq = zx + (size_t)tile * 4 * 256 + (size_t)lane * 8;

  // wait for timestep 0, then load its tile
  while (acq_load(&cnt[0]) < NTILE) __builtin_amdgcn_s_sleep(2);
  v8f zc[4];
#pragma unroll
  for (int g = 0; g < 4; ++g) zc[g] = *(const v8f*)(zq + g * 256);

  for (int s = 0; s < SEQ; ++s) {
    // chase the producers: wait for s+1, then double-buffer its tile
    const int snext = (s + 1 < SEQ) ? (s + 1) : s;
    while (acq_load(&cnt[snext]) < NTILE) __builtin_amdgcn_s_sleep(2);
    const float* zn = zq + (size_t)snext * SSTRIDE;
    v8f zc_next[4];
#pragma unroll
    for (int g = 0; g < 4; ++g) zc_next[g] = *(const v8f*)(zn + g * 256);

    // D_g = h @ Wh_g.T + Zx_g   (one WMMA per gate, K=16 padded to 32)
    v8f d[4];
#pragma unroll
    for (int g = 0; g < 4; ++g)
      d[g] = __builtin_amdgcn_wmma_f32_16x16x32_bf16(false, ah.v, false, whb[g],
                                                     (short)0, zc[g], false, false);

    // elementwise LSTM update in D layout: elem e -> (m = e + 8*hsel, n)
#pragma unroll
    for (int e = 0; e < 8; ++e) {
      float fg = fast_sigmoid(fast_cos(d[0][e]));
      float ig = fast_sigmoid(fast_cos(d[1][e]));
      float ug = fast_tanh   (fast_cos(d[2][e]));
      float og = fast_sigmoid(fast_cos(d[3][e]));
      float cn = __builtin_fmaf(fg, c[e], ig * ug);
      c[e] = cn;
      float hv = og * fast_tanh(cn);
      hsum[e] += hv;
      hlds[(e + hsel * 8) * 16 + n] = (bf16_t)hv;   // row-major store
    }
    __syncthreads();  // single-wave WG: s_barrier -> NOP, just DS ordering
    // A-layout reload: lane reads row r=n, cols k0..k0+7 (contiguous 16B)
    ah.f4[0] = *(const float4*)&hlds[n * 16 + hsel * 8];
    __syncthreads();

#pragma unroll
    for (int g = 0; g < 4; ++g) zc[g] = zc_next[g];
  }

  const float inv = 1.0f / (float)SEQ;
#pragma unroll
  for (int e = 0; e < 8; ++e) {
    int m = e + hsel * 8;
    hmean[(size_t)(tile * 16 + m) * HID + n] = hsum[e] * inv;
  }
}

// ------------------------------ K2: classifier head ------------------------
__global__ void qlstm_head(const float* __restrict__ hmean, const float* __restrict__ Wc,
                           const float* __restrict__ bc, float* __restrict__ out) {
  const int b = threadIdx.x + blockIdx.x * blockDim.x;
  if (b >= BATCH) return;
  float hr[16];
#pragma unroll
  for (int k = 0; k < 16; ++k) hr[k] = hmean[(size_t)b * HID + k];

  float logits[10];
  float mx = -3.4e38f;
#pragma unroll
  for (int cl = 0; cl < 10; ++cl) {
    float a = bc[cl];
#pragma unroll
    for (int k = 0; k < 16; ++k) a = __builtin_fmaf(Wc[cl * 16 + k], hr[k], a);
    logits[cl] = a;
    mx = fmaxf(mx, a);
  }
  float ssum = 0.f;
#pragma unroll
  for (int cl = 0; cl < 10; ++cl)
    ssum += __builtin_amdgcn_exp2f((logits[cl] - mx) * 1.4426950408889634f);
  float lse = __builtin_amdgcn_logf(ssum) * 0.6931471805599453f;
#pragma unroll
  for (int cl = 0; cl < 10; ++cl) out[(size_t)b * 10 + cl] = logits[cl] - mx - lse;
}

// ------------------------------ launcher -----------------------------------
extern "C" void kernel_launch(void* const* d_in, const int* in_sizes, int n_in,
                              void* d_out, int out_size, void* d_ws, size_t ws_size,
                              hipStream_t stream) {
  const float* x   = (const float*)d_in[0];
  const float* Wf  = (const float*)d_in[1];
  const float* bf_ = (const float*)d_in[2];
  const float* thf = (const float*)d_in[3];
  const float* Wi  = (const float*)d_in[4];
  const float* bi_ = (const float*)d_in[5];
  const float* thi = (const float*)d_in[6];
  const float* Wu  = (const float*)d_in[7];
  const float* bu_ = (const float*)d_in[8];
  const float* thu = (const float*)d_in[9];
  const float* Wo  = (const float*)d_in[10];
  const float* bo_ = (const float*)d_in[11];
  const float* tho = (const float*)d_in[12];
  const float* Wc  = (const float*)d_in[13];
  const float* bc  = (const float*)d_in[14];

  char* ws = (char*)d_ws;
  float*  zx    = (float*)ws;                                             // 128 MB
  bf16_t* wtab  = (bf16_t*)(ws + ZX_BYTES);                               // 16 KB
  bf16_t* whtab = (bf16_t*)(ws + ZX_BYTES + 16384);                       // 4 KB
  float*  be    = (float*)(ws + ZX_BYTES + 16384 + 4096);                 // 256 B
  float*  hmean = (float*)(ws + ZX_BYTES + 16384 + 4096 + 256);           // 32 KB
  int*    cnt   = (int*)(ws + ZX_BYTES + 16384 + 4096 + 256 + 32768);     // 4 KB

  qlstm_setup<<<4, 256, 0, stream>>>(Wf, bf_, thf, Wi, bi_, thi, Wu, bu_, thu,
                                     Wo, bo_, tho, wtab, whtab, be, cnt);
  qlstm_fused<<<NTILE + NPROD, 32, 0, stream>>>(x, wtab, whtab, be, zx, cnt, hmean);
  qlstm_head<<<2, 256, 0, stream>>>(hmean, Wc, bc, (float*)d_out);
}
